// attention_prop_13511967113766
// MI455X (gfx1250) — matrix operand order repaired and checked by reading, not compiled, not run on hardware
//
#include <hip/hip_runtime.h>
#include <hip/hip_bf16.h>

#define NROWS 8192
#define FDIM  256

typedef __attribute__((ext_vector_type(16))) __bf16 v16bf;
typedef __attribute__((ext_vector_type(2)))  __bf16 v2bf;
typedef __attribute__((ext_vector_type(8)))  float  v8f;
typedef __attribute__((ext_vector_type(4)))  int    v4i;

#if defined(__has_builtin)
# if __has_builtin(__builtin_amdgcn_global_load_async_to_lds_b128) && \
     __has_builtin(__builtin_amdgcn_s_wait_asynccnt)
#  define HAVE_ASYNC 1
# endif
#endif

#define AS1 __attribute__((address_space(1)))
#define AS3 __attribute__((address_space(3)))

union Frag {
    v16bf v;
    unsigned short s[16];
    unsigned int u[8];
    uint4 q2[2];
};

__device__ __forceinline__ unsigned int pk2bf(float a, float b) {
#if defined(__has_builtin) && __has_builtin(__builtin_amdgcn_cvt_pk_bf16_f32)
    v2bf r = __builtin_amdgcn_cvt_pk_bf16_f32(a, b);       // lo = a, hi = b
#else
    v2bf r = { (__bf16)a, (__bf16)b };                     // native FP_ROUND -> v_cvt if available
#endif
    return __builtin_bit_cast(unsigned int, r);
}

__device__ __forceinline__ unsigned short bf1(float a) {
    __bf16 h = (__bf16)a;
    return __builtin_bit_cast(unsigned short, h);
}

// ---------------------------------------------------------------------------
// Phase 1: h2 = leaky(leaky(x@w1+b1)@w2+b2), store f32 h2 + per-block BN sums
// block = 256 threads (8 waves), 64 rows per block, x is [N,128] f32.
// ---------------------------------------------------------------------------
__global__ __launch_bounds__(256) void mlp_kernel(
    const float* __restrict__ x,
    const float* __restrict__ w1, const float* __restrict__ b1,
    const float* __restrict__ w2, const float* __restrict__ b2,
    float* __restrict__ e_raw,          // [N,256] this branch
    float* __restrict__ psum,           // [gridDim.x,256]
    float* __restrict__ psq)            // [gridDim.x,256]
{
    __shared__ __align__(16) unsigned short sWT[256 * 32];   // weight chunk, transposed [f][k]
    __shared__ __align__(16) unsigned short sH1[64 * 256];   // layer-1 activations bf16

    const int t   = threadIdx.x;
    const int w   = t >> 5;
    const int L   = t & 31;
    const int r   = w >> 1;          // row tile 0..3
    const int ch  = w & 1;           // feature half
    const int lm  = L & 15;
    const int hi  = L >> 4;
    const int row0 = blockIdx.x * 64;

    v8f acc[8] = {};
    Frag a, b;

    // ---- stage A: h1 = leaky(x @ w1 + b1), K = 128 (4 chunks of 32) ----
    for (int kc = 0; kc < 4; ++kc) {
        __syncthreads();
        {   // thread t stages feature row f = t of transposed chunk: sWT[t*32 + 0..31]
            const int k0 = kc * 32;
            unsigned int uu[16];
            for (int j = 0; j < 16; ++j)
                uu[j] = pk2bf(w1[(k0 + 2 * j) * 256 + t], w1[(k0 + 2 * j + 1) * 256 + t]);
            uint4* dst = (uint4*)&sWT[t * 32];
            for (int j = 0; j < 4; ++j)
                dst[j] = uint4{uu[4 * j], uu[4 * j + 1], uu[4 * j + 2], uu[4 * j + 3]};
        }
        __syncthreads();
        {   // A fragment from x (global f32)
            const int row = row0 + 16 * r + lm;
            const int off = kc * 32 + (hi ? 8 : 0);
            const float4* p0 = (const float4*)(x + row * 128 + off);
            const float4* p1 = (const float4*)(x + row * 128 + off + 16);
            float4 u0 = p0[0], u1 = p0[1], v0 = p1[0], v1 = p1[1];
            float g0[8] = {u0.x,u0.y,u0.z,u0.w,u1.x,u1.y,u1.z,u1.w};
            float g1[8] = {v0.x,v0.y,v0.z,v0.w,v1.x,v1.y,v1.z,v1.w};
            for (int j = 0; j < 4; ++j) {
                a.u[j]     = pk2bf(g0[2 * j], g0[2 * j + 1]);
                a.u[4 + j] = pk2bf(g1[2 * j], g1[2 * j + 1]);
            }
        }
        for (int ct = 0; ct < 8; ++ct) {
            const int f  = (ch * 8 + ct) * 16 + lm;
            const int kk = hi ? 8 : 0;
            const uint4* bp = (const uint4*)(&sWT[f * 32 + kk]);
            b.q2[0] = bp[0]; b.q2[1] = bp[2];
            acc[ct] = __builtin_amdgcn_wmma_f32_16x16x32_bf16(
                false, a.v, false, b.v, (short)0, acc[ct], false, false);
        }
    }
    // bias + leaky -> sH1 (bf16)
    for (int ct = 0; ct < 8; ++ct) {
        const int f = (ch * 8 + ct) * 16 + lm;
        const float bias = b1[f];
        for (int vv = 0; vv < 8; ++vv) {
            const int rl = 16 * r + vv + 8 * hi;
            float val = acc[ct][vv] + bias;
            val = (val >= 0.f) ? val : 0.01f * val;
            sH1[rl * 256 + f] = bf1(val);
        }
    }
    {
        v8f z = {};
        for (int i = 0; i < 8; ++i) acc[i] = z;
    }
    __syncthreads();

    // ---- stage B: h2 = leaky(h1 @ w2 + b2), K = 256 (8 chunks of 32) ----
    for (int kc = 0; kc < 8; ++kc) {
        __syncthreads();
        {
            const int k0 = kc * 32;
            unsigned int uu[16];
            for (int j = 0; j < 16; ++j)
                uu[j] = pk2bf(w2[(k0 + 2 * j) * 256 + t], w2[(k0 + 2 * j + 1) * 256 + t]);
            uint4* dst = (uint4*)&sWT[t * 32];
            for (int j = 0; j < 4; ++j)
                dst[j] = uint4{uu[4 * j], uu[4 * j + 1], uu[4 * j + 2], uu[4 * j + 3]};
        }
        __syncthreads();
        {
            const int rl  = 16 * r + lm;
            const int off = kc * 32 + (hi ? 8 : 0);
            a.q2[0] = *(const uint4*)(&sH1[rl * 256 + off]);
            a.q2[1] = *(const uint4*)(&sH1[rl * 256 + off + 16]);
        }
        for (int ct = 0; ct < 8; ++ct) {
            const int f  = (ch * 8 + ct) * 16 + lm;
            const int kk = hi ? 8 : 0;
            const uint4* bp = (const uint4*)(&sWT[f * 32 + kk]);
            b.q2[0] = bp[0]; b.q2[1] = bp[2];
            acc[ct] = __builtin_amdgcn_wmma_f32_16x16x32_bf16(
                false, a.v, false, b.v, (short)0, acc[ct], false, false);
        }
    }
    __syncthreads();   // done reading sWT; reuse as f32 reduction scratch

    float* sRedS = reinterpret_cast<float*>(sWT);   // [256][8]
    float* sRedQ = sRedS + 2048;                    // [256][8]

    for (int ct = 0; ct < 8; ++ct) {
        const int f = (ch * 8 + ct) * 16 + lm;
        const float bias = b2[f];
        float ps = 0.f, pq = 0.f;
        for (int vv = 0; vv < 8; ++vv) {
            const int rl = 16 * r + vv + 8 * hi;
            float val = acc[ct][vv] + bias;
            val = (val >= 0.f) ? val : 0.01f * val;
            e_raw[(size_t)(row0 + rl) * 256 + f] = val;
            ps += val; pq += val * val;
        }
        sRedS[f * 8 + (r * 2 + hi)] = ps;
        sRedQ[f * 8 + (r * 2 + hi)] = pq;
    }
    __syncthreads();
    {   // t == feature index; deterministic 8-way reduction
        float ms = 0.f, mq = 0.f;
        for (int j = 0; j < 8; ++j) { ms += sRedS[t * 8 + j]; mq += sRedQ[t * 8 + j]; }
        psum[blockIdx.x * 256 + t] = ms;
        psq [blockIdx.x * 256 + t] = mq;
    }
}

// ---------------------------------------------------------------------------
// Phase 2a: reduce per-block partials -> mean / inv-std  (grid = 3 branches)
// ---------------------------------------------------------------------------
__global__ __launch_bounds__(256) void bnstats_kernel(
    const float* __restrict__ psum, const float* __restrict__ psq,
    float* __restrict__ meanA, float* __restrict__ istdA)
{
    const int br = blockIdx.x, f = threadIdx.x;
    const float* ps = psum + (size_t)br * 128 * 256;
    const float* pq = psq  + (size_t)br * 128 * 256;
    float ms = 0.f, mq = 0.f;
    for (int i = 0; i < 128; ++i) { ms += ps[i * 256 + f]; mq += pq[i * 256 + f]; }
    const float inv = 1.f / (float)NROWS;
    const float mean = ms * inv;
    const float var  = mq * inv - mean * mean;
    meanA[br * 256 + f] = mean;
    istdA[br * 256 + f] = rsqrtf(var + 1e-5f);
}

// ---------------------------------------------------------------------------
// Phase 2b: apply BN, emit bf16 (packed b32 stores)
// ---------------------------------------------------------------------------
__global__ __launch_bounds__(256) void bnapply_kernel(
    const float* __restrict__ e_raw,
    const float* __restrict__ meanB, const float* __restrict__ istdB,
    const float* __restrict__ gamma, const float* __restrict__ beta,
    unsigned short* __restrict__ ebf)
{
    const int gid  = blockIdx.x * 256 + threadIdx.x;
    const size_t base = (size_t)gid * 8;
    const int f0 = (int)(base & 255);
    unsigned int* eo = (unsigned int*)ebf;
    for (int j = 0; j < 4; ++j) {
        const int fa = f0 + 2 * j, fb = fa + 1;
        const float va = gamma[fa] * (e_raw[base + 2 * j]     - meanB[fa]) * istdB[fa] + beta[fa];
        const float vb = gamma[fb] * (e_raw[base + 2 * j + 1] - meanB[fb]) * istdB[fb] + beta[fb];
        eo[(base >> 1) + j] = pk2bf(va, vb);
    }
}

// ---------------------------------------------------------------------------
// Phase 3: streaming masked attention.
// block = 64 query rows; loop over 128 key tiles of 64 keys.
// A-mask tile staged to LDS with async b128 copies (overlaps S WMMA chain).
// ---------------------------------------------------------------------------
__global__ __launch_bounds__(256) void attn_kernel(
    const unsigned short* __restrict__ qe,
    const unsigned short* __restrict__ ke,
    const unsigned short* __restrict__ ve,
    const float* __restrict__ Am,
    float* __restrict__ out)
{
    __shared__ __align__(16) unsigned short sP [64 * 64];    // P tile bf16
    __shared__ __align__(16) unsigned short sVT[256 * 64];   // ve tile transposed [f][key]
    __shared__ __align__(16) float sA[64 * 64];              // mask tile f32
    __shared__ float sDen[64];

    const int t  = threadIdx.x;
    const int w  = t >> 5;
    const int L  = t & 31;
    const int r  = w >> 1;       // row tile 0..3
    const int ch = w & 1;        // key/feature half
    const int lm = L & 15;
    const int hi = L >> 4;
    const int row0 = blockIdx.x * 64;

    if (t < 64) sDen[t] = 0.f;

    // qe A-fragments for this wave's 16 rows, K = 256 -> 8 chunks (registers)
    Frag qf[8];
    {
        const unsigned short* qrow = qe + (size_t)(row0 + 16 * r + lm) * 256;
        for (int kc = 0; kc < 8; ++kc) {
            const int off = kc * 32 + (hi ? 8 : 0);
            qf[kc].q2[0] = *(const uint4*)(qrow + off);
            qf[kc].q2[1] = *(const uint4*)(qrow + off + 16);
        }
    }

    v8f acc[8] = {};
    float den[8] = {};

    for (int kb = 0; kb < 128; ++kb) {
        const int k0 = kb * 64;

        __syncthreads();   // previous iteration's sP/sVT/sA readers are done

        // ---- stage mask tile A[row0:row0+64, k0:k0+64] -> sA ----
#ifdef HAVE_ASYNC
        for (int i = 0; i < 4; ++i) {
            const int fl  = i * 256 + t;       // uint4 chunk 0..1023
            const int row = fl >> 4;
            const int c4  = fl & 15;
            AS1 v4i* g = (AS1 v4i*)(Am + (size_t)(row0 + row) * NROWS + k0 + c4 * 4);
            AS3 v4i* l = (AS3 v4i*)(&sA[row * 64 + c4 * 4]);
            __builtin_amdgcn_global_load_async_to_lds_b128(g, l, 0, 0);
        }
#else
        for (int i = 0; i < 4; ++i) {
            const int fl  = i * 256 + t;
            const int row = fl >> 4;
            const int c4  = fl & 15;
            *(uint4*)(&sA[row * 64 + c4 * 4]) =
                *(const uint4*)(Am + (size_t)(row0 + row) * NROWS + k0 + c4 * 4);
        }
#endif
        // ---- stage ve tile transposed: sVT[f][key] ----
        {
            const unsigned int* vp = (const unsigned int*)(ve + (size_t)k0 * 256);
            for (int i = 0; i < 32; ++i) {
                int flat = i * 256 + t;       // 0..8191 u32 elems
                int key = flat >> 7;
                int fp  = flat & 127;
                unsigned int u = vp[key * 128 + fp];
                sVT[(2 * fp)     * 64 + key] = (unsigned short)(u & 0xffffu);
                sVT[(2 * fp + 1) * 64 + key] = (unsigned short)(u >> 16);
            }
        }

        // ---- S = qe @ ke^T (overlaps with the async mask copy) ----
        v8f S[2] = {};
        for (int c = 0; c < 2; ++c) {
            const int key = k0 + (ch * 2 + c) * 16 + lm;
            const unsigned short* krow = ke + (size_t)key * 256;
            for (int kc = 0; kc < 8; ++kc) {
                Frag bfr;
                const int off = kc * 32 + (hi ? 8 : 0);
                bfr.q2[0] = *(const uint4*)(krow + off);
                bfr.q2[1] = *(const uint4*)(krow + off + 16);
                S[c] = __builtin_amdgcn_wmma_f32_16x16x32_bf16(
                    false, qf[kc].v, false, bfr.v, (short)0, S[c], false, false);
            }
        }

#ifdef HAVE_ASYNC
        __builtin_amdgcn_s_wait_asynccnt(0);
#endif
        __syncthreads();   // sA + sVT visible to all waves

        // ---- P = exp(A * S / 16); accumulate den; stage P bf16 ----
        for (int c = 0; c < 2; ++c) {
            const int coln = (ch * 2 + c) * 16 + lm;
            for (int vv = 0; vv < 8; ++vv) {
                const int rl = 16 * r + vv + 8 * hi;
                const float av = sA[rl * 64 + coln];
                const float p = __expf(av * S[c][vv] * 0.0625f);
                den[vv] += p;
                sP[rl * 64 + coln] = bf1(p);
            }
        }
        __syncthreads();   // sP visible

        // ---- acc += P @ V (K = 64 keys -> 2 chunks) ----
        Frag pa[2];
        {
            const int rl = 16 * r + lm;
            for (int kc = 0; kc < 2; ++kc) {
                const int off = kc * 32 + (hi ? 8 : 0);
                pa[kc].q2[0] = *(const uint4*)(&sP[rl * 64 + off]);
                pa[kc].q2[1] = *(const uint4*)(&sP[rl * 64 + off + 16]);
            }
        }
        for (int ft = 0; ft < 8; ++ft) {
            const int f = (ch * 8 + ft) * 16 + lm;
            for (int kc = 0; kc < 2; ++kc) {
                Frag bfr;
                const int kk0 = kc * 32 + (hi ? 8 : 0);
                bfr.q2[0] = *(const uint4*)(&sVT[f * 64 + kk0]);
                bfr.q2[1] = *(const uint4*)(&sVT[f * 64 + kk0 + 16]);
                acc[ft] = __builtin_amdgcn_wmma_f32_16x16x32_bf16(
                    false, pa[kc].v, false, bfr.v, (short)0, acc[ft], false, false);
            }
        }
    }

    // ---- softmax denominator: reduce across 16 lanes, combine 2 waves ----
    for (int vv = 0; vv < 8; ++vv) {
        float xsum = den[vv];
        xsum += __shfl_xor(xsum, 1);
        xsum += __shfl_xor(xsum, 2);
        xsum += __shfl_xor(xsum, 4);
        xsum += __shfl_xor(xsum, 8);
        if (lm == 0) atomicAdd(&sDen[16 * r + vv + 8 * hi], xsum);  // exactly 2 adds/row
    }
    __syncthreads();

    for (int ft = 0; ft < 8; ++ft) {
        const int f = (ch * 8 + ft) * 16 + lm;
        for (int vv = 0; vv < 8; ++vv) {
            const int rl = 16 * r + vv + 8 * hi;
            out[(size_t)(row0 + rl) * 256 + f] = acc[ft][vv] / sDen[rl];
        }
    }
}

// ---------------------------------------------------------------------------
extern "C" void kernel_launch(void* const* d_in, const int* in_sizes, int n_in,
                              void* d_out, int out_size, void* d_ws, size_t ws_size,
                              hipStream_t stream) {
    (void)in_sizes; (void)n_in; (void)out_size; (void)ws_size;

    const float* A = (const float*)d_in[0];
    const float* q = (const float*)d_in[1];
    const float* k = (const float*)d_in[2];
    const float* prm[18];
    for (int i = 0; i < 18; ++i) prm[i] = (const float*)d_in[3 + i];

    const size_t NF = (size_t)NROWS * FDIM;
    float* ws    = (float*)d_ws;
    float* psum  = ws;                                  // 3*128*256
    float* psq   = psum + 3 * 128 * 256;                // 3*128*256
    float* meanA = psq  + 3 * 128 * 256;                // 3*256
    float* istdA = meanA + 3 * 256;                     // 3*256
    float* eraw  = istdA + 3 * 256;                     // 3*N*F f32
    unsigned short* ebf = (unsigned short*)(eraw + 3 * NF);  // 3*N*F bf16

    const float* xs[3] = { q, k, k };
    for (int b = 0; b < 3; ++b) {
        mlp_kernel<<<NROWS / 64, 256, 0, stream>>>(
            xs[b], prm[b * 6 + 0], prm[b * 6 + 1], prm[b * 6 + 2], prm[b * 6 + 3],
            eraw + (size_t)b * NF,
            psum + (size_t)b * 128 * 256,
            psq  + (size_t)b * 128 * 256);
    }
    bnstats_kernel<<<3, 256, 0, stream>>>(psum, psq, meanA, istdA);
    for (int b = 0; b < 3; ++b) {
        bnapply_kernel<<<(int)(NF / (256 * 8)), 256, 0, stream>>>(
            eraw + (size_t)b * NF, meanA + b * 256, istdA + b * 256,
            prm[b * 6 + 4], prm[b * 6 + 5], ebf + (size_t)b * NF);
    }
    attn_kernel<<<NROWS / 64, 256, 0, stream>>>(
        ebf, ebf + NF, ebf + 2 * NF, A, (float*)d_out);
}
